// InformerStack_3204045603654
// MI455X (gfx1250) — matrix operand order repaired and checked by reading, not compiled, not run on hardware
//
#include <hip/hip_runtime.h>
#include <cmath>
#include <cstdint>

// ---------------------------------------------------------------------------
// Informer forward for MI455X (gfx1250): LDS-tiled bf16 WMMA GEMM core
// (128x128 block tile, 2x4 fragments/wave, 8 v_wmma per K-step, register
// double-buffered staging, vectorized b128 loads / b64 LDS stores)
// + VALU attention stages.
// ---------------------------------------------------------------------------

#define DM   512
#define NHD  8
#define DH   64
#define DFF  2048
#define BATCH 4

typedef __attribute__((ext_vector_type(16))) __bf16 v16bf;
typedef __attribute__((ext_vector_type(4)))  __bf16 v4bf;
typedef __attribute__((ext_vector_type(8)))  float  v8f;

__device__ __forceinline__ float gelu_f(float x) {
    return 0.5f * x * (1.0f + erff(x * 0.70710678118654752f));
}
__device__ __forceinline__ unsigned wang_hash(unsigned s) {
    s = (s ^ 61u) ^ (s >> 16); s *= 9u; s ^= s >> 4; s *= 0x27d4eb2du; s ^= s >> 15;
    return s;
}
__device__ __forceinline__ float dot4(float4 a, float4 b) {
    return a.x * b.x + a.y * b.y + a.z * b.z + a.w * b.w;
}

// ---------------------------------------------------------------------------
// GEMM: C[M,N] = A[M,K] @ W[K,N] (+bias) (+gelu), fp32 in/out, bf16 WMMA core.
// Block = 256 threads (8 waves) -> 128x128 C tile; wave -> 32x64 = 2x4 frags
// -> 8 v_wmma_f32_16x16x32_bf16 per 32-deep K-step (A frags reused 4x, B 2x).
// Next K-slab is loaded into registers while the current one is consumed from
// LDS; LDS tiles are stored pre-swizzled into WMMA fragment order so fragment
// reads are 2x ds_load_b128 per operand. Global loads are indexed off the
// unmutated kernel-arg pointers (keeps addrspace(1) -> global_load_*).
// Requirements: K % 32 == 0, N % 128 == 0. M ragged OK.
// flags: bit0 = add bias, bit1 = gelu
// ---------------------------------------------------------------------------
__global__ __launch_bounds__(256) void gemm_wmma(
    const float* __restrict__ A, const float* __restrict__ W,
    const float* __restrict__ bias, float* __restrict__ C,
    int M, int N, int K, int flags, int tilesN)
{
    __shared__ __align__(16) __bf16 As[8][32][16];   // 8 KB: 8 M-subtiles
    __shared__ __align__(16) __bf16 Bs[8][32][16];   // 8 KB: 8 N-subtiles

    int tileM = (blockIdx.x / tilesN) << 7;          // *128
    int tileN = (blockIdx.x % tilesN) << 7;          // *128

    int tid  = threadIdx.x;
    int wave = tid >> 5;
    int lane = tid & 31;
    int wm = wave >> 1;        // 0..3 -> M offset wm*32
    int wn = wave & 1;         // 0..1 -> N offset wn*64
    int half = lane >> 4;
    int l16  = lane & 15;

    // ---- per-thread staging offsets (computed once, bumped per K-step) ----
    // A slab 128x32: thread handles 4 float4 quads. quad q = tid + i*256:
    //   row r = q>>3 (0..127), j = (q&7)*4 (j-quad -> one lane, 4 slots)
    size_t  aOff[4];
    __bf16* aDst[4];
    #pragma unroll
    for (int i = 0; i < 4; ++i) {
        int q = tid + i * 256;
        int r = q >> 3, j = (q & 7) * 4;
        int arow = tileM + r; if (arow >= M) arow = M - 1;
        aOff[i] = (size_t)arow * K + j;
        int msub = r >> 4, rr = r & 15;
        int lj = rr + (((j >> 3) & 1) << 4);
        int pj = (j & 7) + ((j >> 4) << 3);
        aDst[i] = &As[msub][lj][pj];
    }
    // B slab 32x128: thread handles 4 (col, j-quad) groups. G = tid + g*256:
    //   c = G & 127, j0 = (G>>7)*4 (j-quad -> one lane, 4 slots)
    size_t  bOff[4];
    __bf16* bDst[4];
    #pragma unroll
    for (int g = 0; g < 4; ++g) {
        int G = tid + g * 256;
        int c = G & 127, j0 = (G >> 7) * 4;
        bOff[g] = (size_t)j0 * N + tileN + c;
        int nsub = c >> 4, cc = c & 15;
        int lj = cc + (((j0 >> 3) & 1) << 4);
        bDst[g] = &Bs[nsub][lj][(j0 & 7) + ((j0 >> 4) << 3)];
    }

    v8f acc[2][4];
    v8f zero = {0.f,0.f,0.f,0.f,0.f,0.f,0.f,0.f};
    #pragma unroll
    for (int i = 0; i < 2; ++i)
        #pragma unroll
        for (int j = 0; j < 4; ++j) acc[i][j] = zero;

    // preload slab 0 into registers
    float4 aV[4];
    float  bV[4][4];
    #pragma unroll
    for (int i = 0; i < 4; ++i) aV[i] = *(const float4*)(A + aOff[i]);
    #pragma unroll
    for (int g = 0; g < 4; ++g)
        #pragma unroll
        for (int t = 0; t < 4; ++t) bV[g][t] = W[bOff[g] + (size_t)t * N];

    for (int k0 = 0; k0 < K; k0 += 32) {
        __syncthreads();   // prior compute finished reading LDS (no-op 1st iter)

        // ---- store current slab regs -> LDS (bf16, fragment order) ----
        #pragma unroll
        for (int i = 0; i < 4; ++i) {
            v4bf pk;
            pk[0] = (__bf16)aV[i].x; pk[1] = (__bf16)aV[i].y;
            pk[2] = (__bf16)aV[i].z; pk[3] = (__bf16)aV[i].w;
            *(v4bf*)aDst[i] = pk;
        }
        #pragma unroll
        for (int g = 0; g < 4; ++g) {
            v4bf pk;
            pk[0] = (__bf16)bV[g][0]; pk[1] = (__bf16)bV[g][1];
            pk[2] = (__bf16)bV[g][2]; pk[3] = (__bf16)bV[g][3];
            *(v4bf*)bDst[g] = pk;
        }

        // ---- issue next slab's global loads (overlap with WMMA below) ----
        bool hasNext = (k0 + 32) < K;
        float4 aN[4];
        float  bN[4][4];
        if (hasNext) {
            #pragma unroll
            for (int i = 0; i < 4; ++i) {
                aOff[i] += 32;
                aN[i] = *(const float4*)(A + aOff[i]);
            }
            #pragma unroll
            for (int g = 0; g < 4; ++g) {
                bOff[g] += (size_t)32 * N;
                #pragma unroll
                for (int t = 0; t < 4; ++t) bN[g][t] = W[bOff[g] + (size_t)t * N];
            }
            // L2 prefetch one more slab ahead
            __builtin_prefetch(A + aOff[0] + 32, 0, 1);
            __builtin_prefetch(W + bOff[0] + (size_t)32 * N, 0, 1);
        }
        __syncthreads();   // LDS stores visible to all waves

        // ---- 8 WMMAs from LDS fragments (2 A-frags x 4 B-frags) ----
        v16bf a0 = *(const v16bf*)As[wm * 2 + 0][lane];
        v16bf a1 = *(const v16bf*)As[wm * 2 + 1][lane];
        v16bf b0 = *(const v16bf*)Bs[wn * 4 + 0][lane];
        v16bf b1 = *(const v16bf*)Bs[wn * 4 + 1][lane];
        v16bf b2 = *(const v16bf*)Bs[wn * 4 + 2][lane];
        v16bf b3 = *(const v16bf*)Bs[wn * 4 + 3][lane];

        acc[0][0] = __builtin_amdgcn_wmma_f32_16x16x32_bf16(false, a0, false, b0, (short)0, acc[0][0], false, false);
        acc[0][1] = __builtin_amdgcn_wmma_f32_16x16x32_bf16(false, a0, false, b1, (short)0, acc[0][1], false, false);
        acc[0][2] = __builtin_amdgcn_wmma_f32_16x16x32_bf16(false, a0, false, b2, (short)0, acc[0][2], false, false);
        acc[0][3] = __builtin_amdgcn_wmma_f32_16x16x32_bf16(false, a0, false, b3, (short)0, acc[0][3], false, false);
        acc[1][0] = __builtin_amdgcn_wmma_f32_16x16x32_bf16(false, a1, false, b0, (short)0, acc[1][0], false, false);
        acc[1][1] = __builtin_amdgcn_wmma_f32_16x16x32_bf16(false, a1, false, b1, (short)0, acc[1][1], false, false);
        acc[1][2] = __builtin_amdgcn_wmma_f32_16x16x32_bf16(false, a1, false, b2, (short)0, acc[1][2], false, false);
        acc[1][3] = __builtin_amdgcn_wmma_f32_16x16x32_bf16(false, a1, false, b3, (short)0, acc[1][3], false, false);

        if (hasNext) {
            #pragma unroll
            for (int i = 0; i < 4; ++i) aV[i] = aN[i];
            #pragma unroll
            for (int g = 0; g < 4; ++g)
                #pragma unroll
                for (int t = 0; t < 4; ++t) bV[g][t] = bN[g][t];
        }
    }

    // ---- store: VGPR r of a fragment -> row (+ half*8 + r), col l16 ----
    #pragma unroll
    for (int i = 0; i < 2; ++i) {
        #pragma unroll
        for (int j = 0; j < 4; ++j) {
            int col = tileN + wn * 64 + j * 16 + l16;
            #pragma unroll
            for (int r = 0; r < 8; ++r) {
                int row = tileM + wm * 32 + i * 16 + half * 8 + r;
                if (row < M) {
                    float v = acc[i][j][r];
                    if (flags & 1) v += bias[col];
                    if (flags & 2) v = gelu_f(v);
                    C[(size_t)row * N + col] = v;
                }
            }
        }
    }
}

// ---------------------------------------------------------------------------
// Data embedding: circular conv1d(k=3) + sinusoid positional + temporal tables
// ---------------------------------------------------------------------------
__global__ void embed_kernel(const float* __restrict__ x, const int* __restrict__ mark,
                             const float* __restrict__ w, const float* __restrict__ bias,
                             float* __restrict__ out, int Bn, int L, int Cin)
{
    size_t i = (size_t)blockIdx.x * blockDim.x + threadIdx.x;
    size_t tot = (size_t)Bn * L * DM;
    if (i >= tot) return;
    int d = (int)(i & (DM - 1));
    size_t bl = i >> 9;
    int l = (int)(bl % L);
    int b = (int)(bl / L);

    float acc = bias[d];
    for (int c = 0; c < Cin; ++c) {
        #pragma unroll
        for (int kk = 0; kk < 3; ++kk) {
            int src = l - 1 + kk;
            src = (src + L) % L;
            acc += w[(d * Cin + c) * 3 + kk] * x[((size_t)b * L + src) * Cin + c];
        }
    }
    int j = d >> 1;
    float div = __expf(-(float)(2 * j) * (9.210340371976184f / 512.0f));
    float pv = (d & 1) ? __cosf((float)l * div) : __sinf((float)l * div);
    float tmp = 0.f;
    #pragma unroll
    for (int m = 0; m < 4; ++m) {
        int mv = mark[((size_t)b * L + l) * 4 + m];
        tmp += (d & 1) ? __cosf((float)mv * div) : __sinf((float)mv * div);
    }
    out[i] = acc + pv + tmp;
}

// h = x[:, L-inp:, :]
__global__ void slice_kernel(const float* __restrict__ x, float* __restrict__ y,
                             int Bn, int L, int inp)
{
    size_t i = (size_t)blockIdx.x * blockDim.x + threadIdx.x;
    size_t tot = (size_t)Bn * inp * DM;
    if (i >= tot) return;
    int d = (int)(i & (DM - 1));
    size_t bl = i >> 9;
    int l = (int)(bl % inp);
    int b = (int)(bl / inp);
    y[i] = x[((size_t)b * L + (L - inp + l)) * DM + d];
}

// dst[b, off+l, :] = src[b, l, :]
__global__ void copy_rows_kernel(const float* __restrict__ src, float* __restrict__ dst,
                                 int Bn, int Lsrc, int Sdst, int off)
{
    size_t i = (size_t)blockIdx.x * blockDim.x + threadIdx.x;
    size_t tot = (size_t)Bn * Lsrc * DM;
    if (i >= tot) return;
    int d = (int)(i & (DM - 1));
    size_t bl = i >> 9;
    int l = (int)(bl % Lsrc);
    int b = (int)(bl / Lsrc);
    dst[((size_t)b * Sdst + off + l) * DM + d] = src[i];
}

// LayerNorm over D=512 (optional residual add first). One block (128 thr) per
// row; each thread owns one float4 (coalesced b128 loads/stores).
__global__ __launch_bounds__(128) void ln_kernel(
    const float* __restrict__ x, const float* __restrict__ res,
    const float* __restrict__ g, const float* __restrict__ bt,
    float* __restrict__ out)
{
    int row = blockIdx.x;
    int t = threadIdx.x;
    const float4* xr = (const float4*)(x + (size_t)row * DM);
    __shared__ float sh[128];
    float4 v = xr[t];
    if (res) {
        float4 r = ((const float4*)(res + (size_t)row * DM))[t];
        v.x += r.x; v.y += r.y; v.z += r.z; v.w += r.w;
    }
    float s = v.x + v.y + v.z + v.w;
    sh[t] = s; __syncthreads();
    for (int o = 64; o > 0; o >>= 1) { if (t < o) sh[t] += sh[t + o]; __syncthreads(); }
    float mean = sh[0] * (1.f / DM); __syncthreads();
    float dx = v.x - mean, dy = v.y - mean, dz = v.z - mean, dw = v.w - mean;
    sh[t] = dx * dx + dy * dy + dz * dz + dw * dw; __syncthreads();
    for (int o = 64; o > 0; o >>= 1) { if (t < o) sh[t] += sh[t + o]; __syncthreads(); }
    float rstd = rsqrtf(sh[0] * (1.f / DM) + 1e-5f);
    float4 gv = ((const float4*)g)[t];
    float4 bv = ((const float4*)bt)[t];
    float4 o4;
    o4.x = dx * rstd * gv.x + bv.x;
    o4.y = dy * rstd * gv.y + bv.y;
    o4.z = dz * rstd * gv.z + bv.z;
    o4.w = dw * rstd * gv.w + bv.w;
    ((float4*)(out + (size_t)row * DM))[t] = o4;
}

// ctx init (non-masked prob attn): ctx[b,l,h,:] = mean_l v
__global__ void ctx_mean_kernel(const float* __restrict__ v, float* __restrict__ ctx,
                                int Bn, int L)
{
    int bh = blockIdx.x; int b = bh >> 3, h = bh & 7;
    int d = threadIdx.x;
    const float* vp = v + (size_t)b * L * DM + h * DH + d;
    float s = 0.f;
    for (int l = 0; l < L; ++l) s += vp[(size_t)l * DM];
    s /= (float)L;
    float* cp = ctx + (size_t)b * L * DM + h * DH + d;
    for (int l = 0; l < L; ++l) cp[(size_t)l * DM] = s;
}

// ctx init (masked prob attn): ctx[b,l,h,:] = cumsum_l v
__global__ void ctx_cumsum_kernel(const float* __restrict__ v, float* __restrict__ ctx,
                                  int Bn, int L)
{
    int bh = blockIdx.x; int b = bh >> 3, h = bh & 7;
    int d = threadIdx.x;
    const float* vp = v + (size_t)b * L * DM + h * DH + d;
    float* cp = ctx + (size_t)b * L * DM + h * DH + d;
    float s = 0.f;
    for (int l = 0; l < L; ++l) { s += vp[(size_t)l * DM]; cp[(size_t)l * DM] = s; }
}

// ProbSparse attention: sampled sparsity measure -> top-u queries ->
// exact softmax attention for selected queries, scatter into ctx.
__global__ __launch_bounds__(256) void prob_attn_kernel(
    const float* __restrict__ q, const float* __restrict__ k, const float* __restrict__ v,
    float* __restrict__ ctx, int L_Q, int L_K, int U_part, int u, int masked, unsigned seed)
{
    __shared__ float Ms[1024];
    __shared__ float prob[1024];
    __shared__ float red[256];
    __shared__ int   redi[256];
    __shared__ int   topidx[64];
    __shared__ __align__(16) float qrow[DH];

    int bh = blockIdx.x; int b = bh >> 3, h = bh & 7;
    const float* qb = q + (size_t)b * L_Q * DM + h * DH;
    const float* kb = k + (size_t)b * L_K * DM + h * DH;
    const float* vb = v + (size_t)b * L_K * DM + h * DH;

    for (int l = threadIdx.x; l < L_Q; l += 256) {
        const float4* qp = (const float4*)(qb + (size_t)l * DM);
        float mx = -1e30f, sum = 0.f;
        for (int s = 0; s < U_part; ++s) {
            unsigned r = wang_hash(seed ^ (unsigned)(l * U_part + s));
            int idx = (int)(r % (unsigned)L_K);
            const float4* kp = (const float4*)(kb + (size_t)idx * DM);
            float dot = 0.f;
            #pragma unroll
            for (int d = 0; d < 16; ++d) dot += dot4(qp[d], kp[d]);
            mx = fmaxf(mx, dot); sum += dot;
        }
        Ms[l] = mx - sum / (float)L_K;
    }
    __syncthreads();

    for (int i = 0; i < u; ++i) {
        float bm = -1e30f; int bi = 0;
        for (int l = threadIdx.x; l < L_Q; l += 256)
            if (Ms[l] > bm) { bm = Ms[l]; bi = l; }
        red[threadIdx.x] = bm; redi[threadIdx.x] = bi; __syncthreads();
        for (int o = 128; o > 0; o >>= 1) {
            if (threadIdx.x < o && red[threadIdx.x + o] > red[threadIdx.x]) {
                red[threadIdx.x] = red[threadIdx.x + o]; redi[threadIdx.x] = redi[threadIdx.x + o];
            }
            __syncthreads();
        }
        if (threadIdx.x == 0) { topidx[i] = redi[0]; Ms[redi[0]] = -1e30f; }
        __syncthreads();
    }

    for (int i = 0; i < u; ++i) {
        int l = topidx[i];
        if (threadIdx.x < DH) qrow[threadIdx.x] = qb[(size_t)l * DM + threadIdx.x];
        __syncthreads();
        const float4* q4 = (const float4*)qrow;
        for (int s = threadIdx.x; s < L_K; s += 256) {
            const float4* kp = (const float4*)(kb + (size_t)s * DM);
            float dot = 0.f;
            #pragma unroll
            for (int d = 0; d < 16; ++d) dot += dot4(q4[d], kp[d]);
            dot *= 0.125f;
            if (masked && s > l) dot = -1e30f;
            prob[s] = dot;
        }
        __syncthreads();
        float mx = -1e30f;
        for (int s = threadIdx.x; s < L_K; s += 256) mx = fmaxf(mx, prob[s]);
        red[threadIdx.x] = mx; __syncthreads();
        for (int o = 128; o > 0; o >>= 1) { if (threadIdx.x < o) red[threadIdx.x] = fmaxf(red[threadIdx.x], red[threadIdx.x + o]); __syncthreads(); }
        mx = red[0]; __syncthreads();
        float sm = 0.f;
        for (int s = threadIdx.x; s < L_K; s += 256) { float e = __expf(prob[s] - mx); prob[s] = e; sm += e; }
        red[threadIdx.x] = sm; __syncthreads();
        for (int o = 128; o > 0; o >>= 1) { if (threadIdx.x < o) red[threadIdx.x] += red[threadIdx.x + o]; __syncthreads(); }
        float inv = 1.f / red[0]; __syncthreads();
        if (threadIdx.x < DH) {
            float acc = 0.f;
            for (int s = 0; s < L_K; ++s) acc += prob[s] * vb[(size_t)s * DM + threadIdx.x];
            ctx[((size_t)b * L_Q + l) * DM + h * DH + threadIdx.x] = acc * inv;
        }
        __syncthreads();
    }
}

// Full attention (decoder cross): one wave per (b,h,l). q row cached in LDS.
__global__ __launch_bounds__(128) void full_attn_kernel(
    const float* __restrict__ q, const float* __restrict__ k, const float* __restrict__ v,
    float* __restrict__ ctx, int Bn, int L, int S)
{
    __shared__ float sc[4][800];
    __shared__ __align__(16) float qsh[4][DH];
    int w = threadIdx.x >> 5;
    int gw = blockIdx.x * 4 + w;
    int tot = Bn * NHD * L;
    if (gw >= tot) gw = tot - 1;
    int lane = threadIdx.x & 31;
    int l = gw % L; int h = (gw / L) & 7; int b = gw / (L * NHD);
    const float* qp = q + ((size_t)b * L + l) * DM + h * DH;
    const float* kb = k + (size_t)b * S * DM + h * DH;
    const float* vb = v + (size_t)b * S * DM + h * DH;
    float* s = sc[w];

    qsh[w][lane] = qp[lane];
    qsh[w][lane + 32] = qp[lane + 32];
    __syncthreads();
    const float4* q4 = (const float4*)qsh[w];

    float mx = -1e30f;
    for (int ss = lane; ss < S; ss += 32) {
        const float4* kp = (const float4*)(kb + (size_t)ss * DM);
        float dot = 0.f;
        #pragma unroll
        for (int d = 0; d < 16; ++d) dot += dot4(q4[d], kp[d]);
        dot *= 0.125f;
        s[ss] = dot; mx = fmaxf(mx, dot);
    }
    __syncthreads();
    for (int o = 16; o > 0; o >>= 1) mx = fmaxf(mx, __shfl_xor(mx, o, 32));
    float sum = 0.f;
    for (int ss = lane; ss < S; ss += 32) { float e = __expf(s[ss] - mx); s[ss] = e; sum += e; }
    __syncthreads();
    for (int o = 16; o > 0; o >>= 1) sum += __shfl_xor(sum, o, 32);
    float inv = 1.f / sum;
    #pragma unroll
    for (int dd = 0; dd < 2; ++dd) {
        int d = lane + dd * 32;
        float acc = 0.f;
        for (int ss = 0; ss < S; ++ss) acc += s[ss] * vb[(size_t)ss * DM + d];
        ctx[((size_t)b * L + l) * DM + h * DH + d] = acc * inv;
    }
}

// "mix" reshape: (B,H,L,Dh) flat -> (B, L, H*Dh)
__global__ void mix_kernel(const float* __restrict__ ctx, float* __restrict__ out,
                           int Bn, int L)
{
    size_t i = (size_t)blockIdx.x * blockDim.x + threadIdx.x;
    size_t tot = (size_t)Bn * L * DM;
    if (i >= tot) return;
    int j = (int)(i & (DM - 1));
    size_t bl = i >> 9;
    int l2 = (int)(bl % L);
    int b = (int)(bl / L);
    size_t f = (size_t)l2 * DM + j;
    int h = (int)(f / ((size_t)L * DH));
    size_t rem = f % ((size_t)L * DH);
    int l = (int)(rem / DH);
    int d = (int)(rem % DH);
    out[i] = ctx[((size_t)b * L + l) * DM + h * DH + d];
}

// im2col for circular conv1d k=3 pad=2 over [B,L,DM] -> [B*(L+2), 1536]
__global__ void im2col_kernel(const float* __restrict__ x, float* __restrict__ col,
                              int Bn, int L)
{
    size_t i = (size_t)blockIdx.x * blockDim.x + threadIdx.x;
    int Lc = L + 2;
    size_t tot = (size_t)Bn * Lc * (3 * DM);
    if (i >= tot) return;
    int kidx = (int)(i % (3 * DM));
    size_t bt = i / (3 * DM);
    int t = (int)(bt % Lc);
    int b = (int)(bt / Lc);
    int c = kidx / 3, kk = kidx % 3;
    int src = t + kk - 2;
    src = ((src % L) + L) % L;
    col[i] = x[((size_t)b * L + src) * DM + c];
}

// w[N=512][K=1536] -> wT[K][N]
__global__ void transpose_w_kernel(const float* __restrict__ w, float* __restrict__ wT)
{
    size_t i = (size_t)blockIdx.x * blockDim.x + threadIdx.x;
    size_t tot = (size_t)DM * 3 * DM;
    if (i >= tot) return;
    int kidx = (int)(i % (3 * DM));
    int n = (int)(i / (3 * DM));
    wT[(size_t)kidx * DM + n] = w[i];
}

// BatchNorm stats over (B, Lc) per channel; y: [B*Lc, DM]
__global__ __launch_bounds__(256) void bn_stats_kernel(
    const float* __restrict__ y, float* __restrict__ stats, int Bn, int Lc)
{
    int d = blockIdx.x;
    __shared__ float sh1[256], sh2[256];
    size_t n = (size_t)Bn * Lc;
    float s = 0.f, s2 = 0.f;
    for (size_t i = threadIdx.x; i < n; i += 256) {
        float t = y[i * DM + d];
        s += t; s2 += t * t;
    }
    sh1[threadIdx.x] = s; sh2[threadIdx.x] = s2; __syncthreads();
    for (int o = 128; o > 0; o >>= 1) {
        if (threadIdx.x < o) { sh1[threadIdx.x] += sh1[threadIdx.x + o]; sh2[threadIdx.x] += sh2[threadIdx.x + o]; }
        __syncthreads();
    }
    if (threadIdx.x == 0) {
        float mean = sh1[0] / (float)n;
        stats[d] = mean;
        stats[DM + d] = sh2[0] / (float)n - mean * mean;
    }
}

// BN-normalize + ELU + maxpool(k=3, s=2, pad=1): y[B*Lc,DM] -> out[B*Lp,DM]
__global__ void bn_elu_pool_kernel(const float* __restrict__ y, const float* __restrict__ stats,
                                   const float* __restrict__ g, const float* __restrict__ be,
                                   float* __restrict__ out, int Bn, int Lc, int Lp)
{
    size_t i = (size_t)blockIdx.x * blockDim.x + threadIdx.x;
    size_t tot = (size_t)Bn * Lp * DM;
    if (i >= tot) return;
    int d = (int)(i & (DM - 1));
    size_t bl = i >> 9;
    int tp = (int)(bl % Lp);
    int b = (int)(bl / Lp);
    float mean = stats[d];
    float rstd = rsqrtf(stats[DM + d] + 1e-5f);
    float mx = -1e30f;
    #pragma unroll
    for (int kk = 0; kk < 3; ++kk) {
        int t = 2 * tp - 1 + kk;
        if (t >= 0 && t < Lc) {
            float val = (y[((size_t)b * Lc + t) * DM + d] - mean) * rstd * g[d] + be[d];
            val = val > 0.f ? val : (__expf(val) - 1.f);
            mx = fmaxf(mx, val);
        }
    }
    out[i] = mx;
}

// final projection (N=7) + PRED_LEN slice
__global__ void proj_kernel(const float* __restrict__ x, const float* __restrict__ w,
                            const float* __restrict__ bias, float* __restrict__ out,
                            int Bn, int L, int predOff, int predLen)
{
    size_t i = (size_t)blockIdx.x * blockDim.x + threadIdx.x;
    size_t tot = (size_t)Bn * predLen * 7;
    if (i >= tot) return;
    int c = (int)(i % 7);
    int t = (int)((i / 7) % predLen);
    int b = (int)(i / (7 * predLen));
    const float4* x4 = (const float4*)(x + ((size_t)b * L + predOff + t) * DM);
    float acc = bias[c];
    for (int d4 = 0; d4 < 128; ++d4) {
        float4 xv = x4[d4];
        int d = d4 * 4;
        acc += xv.x * w[d * 7 + c] + xv.y * w[(d + 1) * 7 + c]
             + xv.z * w[(d + 2) * 7 + c] + xv.w * w[(d + 3) * 7 + c];
    }
    out[i] = acc;
}

// ---------------------------------------------------------------------------
// Host orchestration
// ---------------------------------------------------------------------------
struct AttnP { const float *wq,*bq,*wk,*bk,*wv,*bv,*wo,*bo; };
struct EncLayerP { AttnP attn; const float *w1,*b1,*w2,*b2,*n1g,*n1b,*n2g,*n2b; };
struct ConvP { const float *w,*b,*g,*be; };
struct DecLayerP { AttnP sa, ca; const float *w1,*b1,*w2,*b2,*n1g,*n1b,*n2g,*n2b,*n3g,*n3b; };

static inline int cdiv(int a, int b) { return (a + b - 1) / b; }
static inline int ufac(int L) {
    int c = (int)std::ceil(std::log((double)L));
    int v = 5 * c; return v < L ? v : L;
}

static void launch_gemm(const float* A, const float* W, const float* bias, float* C,
                        int M, int N, int K, int flags, hipStream_t stream)
{
    int tilesM = cdiv(M, 128);
    int tilesN = N / 128;
    gemm_wmma<<<tilesM * tilesN, 256, 0, stream>>>(A, W, bias, C, M, N, K, flags, tilesN);
}

extern "C" void kernel_launch(void* const* d_in, const int* in_sizes, int n_in,
                              void* d_out, int out_size, void* d_ws, size_t ws_size,
                              hipStream_t stream)
{
    (void)in_sizes; (void)n_in; (void)out_size; (void)ws_size;

    const int SEQ = 1024, DECL = 768, PRED = 256;

    const float* x_enc      = (const float*)d_in[0];
    const int*   x_mark_enc = (const int*)  d_in[1];
    const float* x_dec      = (const float*)d_in[2];
    const int*   x_mark_dec = (const int*)  d_in[3];

    int ci = 4;
    auto nx = [&]() { return (const float*)d_in[ci++]; };

    const float *enc_emb_w, *enc_emb_b, *dec_emb_w, *dec_emb_b;
    enc_emb_w = nx(); enc_emb_b = nx();
    dec_emb_w = nx(); dec_emb_b = nx();

    const int EL[3] = {3, 2, 1};
    EncLayerP encL[3][3];
    ConvP     encC[3][2];
    const float* encNg[3]; const float* encNb[3];
    for (int s = 0; s < 3; ++s) {
        for (int li = 0; li < EL[s]; ++li) {
            AttnP a;
            a.wq = nx(); a.bq = nx(); a.wk = nx(); a.bk = nx();
            a.wv = nx(); a.bv = nx(); a.wo = nx(); a.bo = nx();
            EncLayerP& e = encL[s][li];
            e.attn = a;
            e.w1 = nx(); e.b1 = nx(); e.w2 = nx(); e.b2 = nx();
            e.n1g = nx(); e.n1b = nx(); e.n2g = nx(); e.n2b = nx();
        }
        for (int cc = 0; cc < EL[s] - 1; ++cc) {
            ConvP& c = encC[s][cc];
            c.w = nx(); c.b = nx(); c.g = nx(); c.be = nx();
        }
        encNg[s] = nx(); encNb[s] = nx();
    }
    DecLayerP decL[2];
    for (int li = 0; li < 2; ++li) {
        DecLayerP& dl = decL[li];
        dl.sa.wq = nx(); dl.sa.bq = nx(); dl.sa.wk = nx(); dl.sa.bk = nx();
        dl.sa.wv = nx(); dl.sa.bv = nx(); dl.sa.wo = nx(); dl.sa.bo = nx();
        dl.ca.wq = nx(); dl.ca.bq = nx(); dl.ca.wk = nx(); dl.ca.bk = nx();
        dl.ca.wv = nx(); dl.ca.bv = nx(); dl.ca.wo = nx(); dl.ca.bo = nx();
        dl.w1 = nx(); dl.b1 = nx(); dl.w2 = nx(); dl.b2 = nx();
        dl.n1g = nx(); dl.n1b = nx(); dl.n2g = nx(); dl.n2b = nx();
        dl.n3g = nx(); dl.n3b = nx();
    }
    const float* dec_ng = nx(); const float* dec_nb = nx();
    const float* proj_w = nx(); const float* proj_b = nx();

    float* base = (float*)d_ws;
    size_t off = 0;
    auto alloc = [&](size_t n) { float* p = base + off; off += (n + 63) & ~(size_t)63; return p; };
    float* ENCX  = alloc((size_t)BATCH * SEQ * DM);
    float* H     = alloc((size_t)BATCH * SEQ * DM);
    float* Q     = alloc((size_t)BATCH * SEQ * DM);
    float* Kb    = alloc((size_t)BATCH * SEQ * DM);
    float* Vb    = alloc((size_t)BATCH * SEQ * DM);
    float* CTX   = alloc((size_t)BATCH * SEQ * DM);
    float* TMPA  = alloc((size_t)BATCH * SEQ * DM);
    float* TMPB  = alloc((size_t)BATCH * SEQ * DM);
    float* FFbuf = alloc((size_t)BATCH * SEQ * DFF);
    float* COL   = alloc((size_t)BATCH * (SEQ + 2) * 3 * DM);
    float* CONVY = alloc((size_t)BATCH * (SEQ + 2) * DM);
    float* WT    = alloc((size_t)3 * DM * DM);
    float* STATS = alloc(2 * DM);
    float* ENCOUT= alloc((size_t)BATCH * 800 * DM);
    float* DECX  = alloc((size_t)BATCH * DECL * DM);

    int stackLen[3];
    for (int s = 0; s < 3; ++s) {
        int L = SEQ >> s;
        for (int li = 0; li < EL[s] - 1; ++li) { int Lc = L + 2; L = (Lc - 1) / 2 + 1; }
        stackLen[s] = L;
    }
    int Stot = stackLen[0] + stackLen[1] + stackLen[2];

    {
        size_t tot = (size_t)BATCH * SEQ * DM;
        embed_kernel<<<cdiv((int)tot, 256), 256, 0, stream>>>(
            x_enc, x_mark_enc, enc_emb_w, enc_emb_b, ENCX, BATCH, SEQ, 7);
        tot = (size_t)BATCH * DECL * DM;
        embed_kernel<<<cdiv((int)tot, 256), 256, 0, stream>>>(
            x_dec, x_mark_dec, dec_emb_w, dec_emb_b, DECX, BATCH, DECL, 7);
    }

    int Soff = 0;
    for (int s = 0; s < 3; ++s) {
        int L = SEQ >> s;
        {
            size_t tot = (size_t)BATCH * L * DM;
            slice_kernel<<<cdiv((int)tot, 256), 256, 0, stream>>>(ENCX, H, BATCH, SEQ, L);
        }
        for (int li = 0; li < EL[s]; ++li) {
            const EncLayerP& lp = encL[s][li];
            int M = BATCH * L;
            unsigned seed = 1000u + (unsigned)(s * 10 + li);
            launch_gemm(H, lp.attn.wq, lp.attn.bq, Q,  M, DM, DM, 1, stream);
            launch_gemm(H, lp.attn.wk, lp.attn.bk, Kb, M, DM, DM, 1, stream);
            launch_gemm(H, lp.attn.wv, lp.attn.bv, Vb, M, DM, DM, 1, stream);
            ctx_mean_kernel<<<BATCH * NHD, DH, 0, stream>>>(Vb, CTX, BATCH, L);
            int U = ufac(L), u = ufac(L);
            prob_attn_kernel<<<BATCH * NHD, 256, 0, stream>>>(Q, Kb, Vb, CTX, L, L, U, u, 0, seed);
            launch_gemm(CTX, lp.attn.wo, lp.attn.bo, TMPA, M, DM, DM, 1, stream);
            ln_kernel<<<M, 128, 0, stream>>>(H, TMPA, lp.n1g, lp.n1b, TMPB);
            launch_gemm(TMPB, lp.w1, lp.b1, FFbuf, M, DFF, DM, 1 | 2, stream);
            launch_gemm(FFbuf, lp.w2, lp.b2, TMPA, M, DM, DFF, 1, stream);
            ln_kernel<<<M, 128, 0, stream>>>(TMPB, TMPA, lp.n2g, lp.n2b, H);
            if (li < EL[s] - 1) {
                const ConvP& cp = encC[s][li];
                {
                    size_t tot = (size_t)3 * DM * DM;
                    transpose_w_kernel<<<cdiv((int)tot, 256), 256, 0, stream>>>(cp.w, WT);
                }
                int Lc = L + 2, McT = BATCH * Lc;
                {
                    size_t tot = (size_t)McT * 3 * DM;
                    im2col_kernel<<<cdiv((int)tot, 256), 256, 0, stream>>>(H, COL, BATCH, L);
                }
                launch_gemm(COL, WT, cp.b, CONVY, McT, DM, 3 * DM, 1, stream);
                bn_stats_kernel<<<DM, 256, 0, stream>>>(CONVY, STATS, BATCH, Lc);
                int Lp = (Lc - 1) / 2 + 1;
                {
                    size_t tot = (size_t)BATCH * Lp * DM;
                    bn_elu_pool_kernel<<<cdiv((int)tot, 256), 256, 0, stream>>>(
                        CONVY, STATS, cp.g, cp.be, H, BATCH, Lc, Lp);
                }
                L = Lp;
            }
        }
        int M = BATCH * L;
        ln_kernel<<<M, 128, 0, stream>>>(H, nullptr, encNg[s], encNb[s], TMPA);
        {
            size_t tot = (size_t)M * DM;
            copy_rows_kernel<<<cdiv((int)tot, 256), 256, 0, stream>>>(TMPA, ENCOUT, BATCH, L, Stot, Soff);
        }
        Soff += L;
    }

    int Md = BATCH * DECL;
    int Mc = BATCH * Stot;
    for (int li = 0; li < 2; ++li) {
        const DecLayerP& dl = decL[li];
        unsigned seed = 2000u + (unsigned)li;
        launch_gemm(DECX, dl.sa.wq, dl.sa.bq, Q,  Md, DM, DM, 1, stream);
        launch_gemm(DECX, dl.sa.wk, dl.sa.bk, Kb, Md, DM, DM, 1, stream);
        launch_gemm(DECX, dl.sa.wv, dl.sa.bv, Vb, Md, DM, DM, 1, stream);
        ctx_cumsum_kernel<<<BATCH * NHD, DH, 0, stream>>>(Vb, CTX, BATCH, DECL);
        int u = ufac(DECL);
        prob_attn_kernel<<<BATCH * NHD, 256, 0, stream>>>(Q, Kb, Vb, CTX, DECL, DECL, u, u, 1, seed);
        {
            size_t tot = (size_t)Md * DM;
            mix_kernel<<<cdiv((int)tot, 256), 256, 0, stream>>>(CTX, TMPA, BATCH, DECL);
        }
        launch_gemm(TMPA, dl.sa.wo, dl.sa.bo, TMPB, Md, DM, DM, 1, stream);
        ln_kernel<<<Md, 128, 0, stream>>>(DECX, TMPB, dl.n1g, dl.n1b, DECX);
        launch_gemm(DECX,  dl.ca.wq, dl.ca.bq, Q,  Md, DM, DM, 1, stream);
        launch_gemm(ENCOUT, dl.ca.wk, dl.ca.bk, Kb, Mc, DM, DM, 1, stream);
        launch_gemm(ENCOUT, dl.ca.wv, dl.ca.bv, Vb, Mc, DM, DM, 1, stream);
        {
            int totw = BATCH * NHD * DECL;
            full_attn_kernel<<<cdiv(totw, 4), 128, 0, stream>>>(Q, Kb, Vb, CTX, BATCH, DECL, Stot);
        }
        launch_gemm(CTX, dl.ca.wo, dl.ca.bo, TMPB, Md, DM, DM, 1, stream);
        ln_kernel<<<Md, 128, 0, stream>>>(DECX, TMPB, dl.n2g, dl.n2b, DECX);
        launch_gemm(DECX, dl.w1, dl.b1, FFbuf, Md, DFF, DM, 1 | 2, stream);
        launch_gemm(FFbuf, dl.w2, dl.b2, TMPA, Md, DM, DFF, 1, stream);
        ln_kernel<<<Md, 128, 0, stream>>>(DECX, TMPA, dl.n3g, dl.n3b, DECX);
    }
    ln_kernel<<<Md, 128, 0, stream>>>(DECX, nullptr, dec_ng, dec_nb, TMPA);
    {
        size_t tot = (size_t)BATCH * PRED * 7;
        proj_kernel<<<cdiv((int)tot, 256), 256, 0, stream>>>(
            TMPA, proj_w, proj_b, (float*)d_out, BATCH, DECL, DECL - PRED, PRED);
    }
}
